// RelationalConceptGraphLearning_34007551050434
// MI455X (gfx1250) — matrix operand order repaired and checked by reading, not compiled, not run on hardware
//
#include <hip/hip_runtime.h>
#include <hip/hip_bf16.h>
#include <math.h>

typedef __attribute__((ext_vector_type(16))) _Float16 v16h;
typedef __attribute__((ext_vector_type(8)))  _Float16 v8h;
typedef __attribute__((ext_vector_type(8)))  float    v8f;

__device__ __forceinline__ v16h cat8(v8h lo, v8h hi) {
    return __builtin_shufflevector(lo, hi, 0,1,2,3,4,5,6,7,8,9,10,11,12,13,14,15);
}
__device__ __forceinline__ v8f wmma16(v16h a, v16h b, v8f c) {
    return __builtin_amdgcn_wmma_f32_16x16x32_f16(false, a, false, b, (short)0, c, false, false);
}
__device__ __forceinline__ uint32_t pack2h(float a, float b) {
    union { _Float16 hh[2]; uint32_t u; } pk;
    pk.hh[0] = (_Float16)a; pk.hh[1] = (_Float16)b;
    return pk.u;
}

// ---------------------------------------------------------------------------
// f16-WMMA GEMM: out[M,N] = act(A[M,K] @ W[K,N] + bias[N])
// 256 threads (8 waves), 256x64 tile. Each wave: 2 row strips x 4 col tiles.
// B tile stored TRANSPOSED in LDS so fragments are two contiguous 16B loads.
// Staging: guard-free packed-dword fast path; compact guarded tail (K=500).
// ---------------------------------------------------------------------------
__global__ __launch_bounds__(256)
void gemm_xw_kernel(const float* __restrict__ A, const float* __restrict__ W,
                    const float* __restrict__ bias, float* __restrict__ out,
                    int M, int N, int K, int relu)
{
    __shared__ __align__(16) _Float16 As[256][40];   // 256x32 (+8 pad, 80B stride)
    __shared__ __align__(16) _Float16 BsT[64][40];   // transposed: [col][k]

    const int tid  = threadIdx.x;
    const int lane = tid & 31;
    const int wave = tid >> 5;                  // 0..7
    const int row0 = blockIdx.x * 256;
    const int col0 = blockIdx.y * 64;

    v8f acc[2][4] = {};

    const int arow0 = wave * 16 + (lane & 15);  // strip 0 row
    const int arow1 = 128 + arow0;              // strip 1 row
    const int koff  = (lane >> 4) << 3;         // 0 / 8
    const int bcol  = lane & 15;
    const int kb    = (lane >> 4) << 3;         // 0 / 8

    for (int k0 = 0; k0 < K; k0 += 32) {
        if (k0 + 32 <= K) {
            // ---- fast path: no guards, packed dword LDS stores ----
            #pragma unroll
            for (int i = 0; i < 16; ++i) {
                int p  = i * 256 + tid;          // 0..4095 (pairs)
                int r  = p >> 4;                 // 0..255
                int cp = (p & 15) * 2;           // even col
                const float2 v2 = *(const float2*)&A[(size_t)(row0 + r) * K + (k0 + cp)];
                *(uint32_t*)&As[r][cp] = pack2h(v2.x, v2.y);
            }
            #pragma unroll
            for (int i = 0; i < 4; ++i) {
                int p  = i * 256 + tid;          // 0..1023 (pairs along k)
                int c  = p & 63;
                int rp = (p >> 6) * 2;           // even k
                float w0 = W[(size_t)(k0 + rp) * N + (col0 + c)];
                float w1 = W[(size_t)(k0 + rp + 1) * N + (col0 + c)];
                *(uint32_t*)&BsT[c][rp] = pack2h(w0, w1);
            }
            if (k0 + 32 < K) {   // prefetch next k-tile into caches
                __builtin_prefetch(&A[(size_t)(row0 + (tid >> 4)) * K + (k0 + 32)], 0, 1);
                __builtin_prefetch(&W[(size_t)(k0 + 32 + (tid >> 6)) * N + (col0 + (tid & 63))], 0, 1);
            }
        } else {
            // ---- guarded tail (runs once for K=500), keep compact ----
            #pragma unroll 1
            for (int e = tid; e < 8192; e += 256) {
                int r = e >> 5, c = e & 31;
                float v = (k0 + c < K) ? A[(size_t)(row0 + r) * K + (k0 + c)] : 0.0f;
                As[r][c] = (_Float16)v;
            }
            #pragma unroll 1
            for (int e = tid; e < 2048; e += 256) {
                int r = e >> 6, c = e & 63;
                float v = (k0 + r < K) ? W[(size_t)(k0 + r) * N + (col0 + c)] : 0.0f;
                BsT[c][r] = (_Float16)v;
            }
        }
        __syncthreads();

        v16h a0 = cat8(*(const v8h*)&As[arow0][koff], *(const v8h*)&As[arow0][16 + koff]);
        v16h a1 = cat8(*(const v8h*)&As[arow1][koff], *(const v8h*)&As[arow1][16 + koff]);
        #pragma unroll
        for (int jt = 0; jt < 4; ++jt) {
            const _Float16* bp = &BsT[jt * 16 + bcol][kb];
            v16h bf = cat8(*(const v8h*)bp, *(const v8h*)(bp + 16));
            acc[0][jt] = wmma16(a0, bf, acc[0][jt]);
            acc[1][jt] = wmma16(a1, bf, acc[1][jt]);
        }
        __syncthreads();
    }

    const int ccol  = lane & 15;
    const int rbase = (lane >> 4) << 3;
    #pragma unroll
    for (int s = 0; s < 2; ++s) {
        #pragma unroll
        for (int jt = 0; jt < 4; ++jt) {
            #pragma unroll
            for (int v = 0; v < 8; ++v) {
                int r = row0 + s * 128 + wave * 16 + rbase + v;
                int c = col0 + jt * 16 + ccol;
                float val = acc[s][jt][v] + bias[c];
                if (relu) val = fmaxf(val, 0.0f);
                out[(size_t)r * N + c] = val;
            }
        }
    }
}

// ---------------------------------------------------------------------------
// Per-(mat,b,h) LayerNorm over [N=128, NODE=64] with affine, emit f16.
// proj: [3][32768][512] fp32 ; qkv: [3][256][8][128*64] f16
// ---------------------------------------------------------------------------
__global__ __launch_bounds__(256)
void headnorm_kernel(const float* __restrict__ proj,
                     const float* __restrict__ kn_w, const float* __restrict__ kn_b,
                     const float* __restrict__ qn_w, const float* __restrict__ qn_b,
                     const float* __restrict__ vn_w, const float* __restrict__ vn_b,
                     _Float16* __restrict__ qkv)
{
    __shared__ float red[256];
    __shared__ float red2[256];

    const int bi  = blockIdx.x;              // mat*2048 + b*8 + h
    const int mat = bi >> 11;
    const int b   = (bi >> 3) & 255;
    const int h   = bi & 7;
    const int tid = threadIdx.x;

    const float* w  = (mat == 0) ? kn_w : (mat == 1) ? qn_w : vn_w;
    const float* bb = (mat == 0) ? kn_b : (mat == 1) ? qn_b : vn_b;
    const float* src = proj + ((size_t)mat * 32768 + (size_t)b * 128) * 512 + h * 64;

    float s = 0.0f, s2 = 0.0f;
    #pragma unroll 1
    for (int e = tid; e < 8192; e += 256) {
        int n = e >> 6, d = e & 63;
        float v = src[n * 512 + d];
        s += v; s2 += v * v;
    }
    red[tid] = s; red2[tid] = s2;
    __syncthreads();
    for (int off = 128; off > 0; off >>= 1) {
        if (tid < off) { red[tid] += red[tid + off]; red2[tid] += red2[tid + off]; }
        __syncthreads();
    }
    const float mean = red[0] * (1.0f / 8192.0f);
    const float var  = red2[0] * (1.0f / 8192.0f) - mean * mean;
    const float rs   = rsqrtf(var + 1e-5f);

    // Write packed pairs (one dword per 2 elements)
    uint32_t* dst = (uint32_t*)(qkv + ((size_t)mat * 2048 + (size_t)b * 8 + h) * 8192);
    #pragma unroll 1
    for (int p = tid; p < 4096; p += 256) {
        int e = p * 2;
        int n = e >> 6, d = e & 63;
        const float2 v2 = *(const float2*)&src[n * 512 + d];
        dst[p] = pack2h((v2.x - mean) * rs * w[e] + bb[e],
                        (v2.y - mean) * rs * w[e + 1] + bb[e + 1]);
    }
}

// ---------------------------------------------------------------------------
// Fused attention per (b,h): S = Q K^T, softmax(S/8) -> A (d_out fp32 + LDS f16),
// E = A V (merged-head fp32 to workspace). WMMA with wide b128 LDS fragments.
// ---------------------------------------------------------------------------
__global__ __launch_bounds__(256)
void attn_kernel(const _Float16* __restrict__ qkv,  // [3][256][8][8192]
                 float* __restrict__ A_out,         // [256][8][128][128]
                 float* __restrict__ E_out)         // [32768][512]
{
    __shared__ __align__(16) _Float16 Qs[128][64];    // 16 KB
    __shared__ __align__(16) _Float16 Ks[128][64];    // 16 KB ([n][d] == (K^T) col-major)
    __shared__ __align__(16) _Float16 VsT[64][128];   // 16 KB, transposed [d][n]
    __shared__ __align__(16) float    Ss[128][128];   // 64 KB
    __shared__ __align__(16) _Float16 Af[128][128];   // 32 KB, softmaxed A in f16

    const int bh   = blockIdx.x;     // 0..2047
    const int b    = bh >> 3;
    const int h    = bh & 7;
    const int tid  = threadIdx.x;
    const int lane = tid & 31;
    const int wave = tid >> 5;       // 0..7

    const size_t base      = (size_t)bh * 8192;
    const size_t matStride = (size_t)2048 * 8192;

    // Stage Q,K as 16B chunks
    const uint4* gK4 = (const uint4*)(qkv + 0 * matStride + base);
    const uint4* gQ4 = (const uint4*)(qkv + 1 * matStride + base);
    uint4* sQ4 = (uint4*)&Qs[0][0];
    uint4* sK4 = (uint4*)&Ks[0][0];
    #pragma unroll 1
    for (int i = tid; i < 1024; i += 256) { sQ4[i] = gQ4[i]; sK4[i] = gK4[i]; }
    // Stage V transposed: VsT[d][n]
    const _Float16* gV = qkv + 2 * matStride + base;
    #pragma unroll
    for (int g = 0; g < 4; ++g) {
        int e = (g * 256 + tid) * 8;          // 16B-aligned group of 8 halves
        int n = e >> 6, d = e & 63;
        v8h v = *(const v8h*)(gV + e);
        #pragma unroll
        for (int j = 0; j < 8; ++j) VsT[d + j][n] = v[j];
    }
    __syncthreads();

    const int arow  = wave * 16 + (lane & 15);
    const int koff  = (lane >> 4) << 3;
    const int bcol  = lane & 15;
    const int kb    = (lane >> 4) << 3;
    const int ccol  = lane & 15;
    const int rbase = (lane >> 4) << 3;

    // ---- S = Q @ K^T : M=128, N=128, K=64 ----
    v8f sacc[8] = {};
    #pragma unroll
    for (int k0 = 0; k0 < 64; k0 += 32) {
        v16h af = cat8(*(const v8h*)&Qs[arow][k0 + koff],
                       *(const v8h*)&Qs[arow][k0 + 16 + koff]);
        #pragma unroll
        for (int jt = 0; jt < 8; ++jt) {
            const _Float16* bp = &Ks[jt * 16 + bcol][k0 + kb];   // (K^T)[k][n]=K[n][k]
            v16h bf = cat8(*(const v8h*)bp, *(const v8h*)(bp + 16));
            sacc[jt] = wmma16(af, bf, sacc[jt]);
        }
    }
    #pragma unroll
    for (int jt = 0; jt < 8; ++jt)
        #pragma unroll
        for (int v = 0; v < 8; ++v)
            Ss[wave * 16 + rbase + v][jt * 16 + ccol] = sacc[jt][v];
    __syncthreads();

    // ---- row softmax of S/8 -> A (fp32 to d_out, f16 to Af) ----
    if (tid < 128) {
        float* srow = &Ss[tid][0];
        float m = -INFINITY;
        for (int c = 0; c < 128; ++c) m = fmaxf(m, srow[c]);
        float sum = 0.0f;
        for (int c = 0; c < 128; ++c) {
            float p = __expf((srow[c] - m) * 0.125f);
            srow[c] = p; sum += p;
        }
        const float inv = 1.0f / sum;
        float*    arow_out = A_out + ((size_t)bh * 128 + tid) * 128;
        uint32_t* afrow    = (uint32_t*)&Af[tid][0];
        for (int c = 0; c < 128; c += 2) {
            float a0 = srow[c] * inv, a1 = srow[c + 1] * inv;
            arow_out[c] = a0; arow_out[c + 1] = a1;
            afrow[c >> 1] = pack2h(a0, a1);
        }
    }
    __syncthreads();

    // ---- E = A @ V : M=128, N=64, K=128 ----
    v8f eacc[4] = {};
    #pragma unroll
    for (int k0 = 0; k0 < 128; k0 += 32) {
        v16h af = cat8(*(const v8h*)&Af[arow][k0 + koff],
                       *(const v8h*)&Af[arow][k0 + 16 + koff]);
        #pragma unroll
        for (int jt = 0; jt < 4; ++jt) {
            const _Float16* bp = &VsT[jt * 16 + bcol][k0 + kb];
            v16h bf = cat8(*(const v8h*)bp, *(const v8h*)(bp + 16));
            eacc[jt] = wmma16(af, bf, eacc[jt]);
        }
    }
    #pragma unroll
    for (int jt = 0; jt < 4; ++jt)
        #pragma unroll
        for (int v = 0; v < 8; ++v) {
            int n = wave * 16 + rbase + v;
            int d = jt * 16 + ccol;
            E_out[((size_t)b * 128 + n) * 512 + h * 64 + d] = eacc[jt][v];
        }
}

// ---------------------------------------------------------------------------
// LN over (N,HID) per batch (no affine) + maxpool over N. F:[256][128][512]
// ---------------------------------------------------------------------------
__global__ __launch_bounds__(256)
void finalnorm_maxpool_kernel(const float* __restrict__ F, float* __restrict__ G)
{
    __shared__ float red[256];
    __shared__ float red2[256];
    const int b = blockIdx.x, tid = threadIdx.x;
    const float* src = F + (size_t)b * 65536;

    float s = 0.0f, s2 = 0.0f;
    #pragma unroll 1
    for (int e = tid; e < 65536; e += 256) { float v = src[e]; s += v; s2 += v * v; }
    red[tid] = s; red2[tid] = s2;
    __syncthreads();
    for (int off = 128; off > 0; off >>= 1) {
        if (tid < off) { red[tid] += red[tid + off]; red2[tid] += red2[tid + off]; }
        __syncthreads();
    }
    const float mean = red[0] * (1.0f / 65536.0f);
    const float var  = red2[0] * (1.0f / 65536.0f) - mean * mean;
    const float rs   = rsqrtf(var + 1e-5f);

    #pragma unroll 1
    for (int c = tid; c < 512; c += 256) {
        float m = -INFINITY;
        for (int n = 0; n < 128; ++n)
            m = fmaxf(m, (src[n * 512 + c] - mean) * rs);
        G[(size_t)b * 512 + c] = m;
    }
}

// ---------------------------------------------------------------------------
// y = elu(G @ W2 + b2). G:[256][512], W2:[512][256]. Tiny: VALU dot.
// ---------------------------------------------------------------------------
__global__ __launch_bounds__(256)
void head_out_kernel(const float* __restrict__ G, const float* __restrict__ W2,
                     const float* __restrict__ b2, float* __restrict__ y)
{
    const int b = blockIdx.x, o = threadIdx.x;
    float s = b2[o];
    #pragma unroll 1
    for (int k = 0; k < 512; ++k)
        s = fmaf(G[(size_t)b * 512 + k], W2[(size_t)k * 256 + o], s);
    y[(size_t)b * 256 + o] = (s > 0.0f) ? s : expm1f(s);
}

// ---------------------------------------------------------------------------
extern "C" void kernel_launch(void* const* d_in, const int* in_sizes, int n_in,
                              void* d_out, int out_size, void* d_ws, size_t ws_size,
                              hipStream_t stream)
{
    const float* x    = (const float*)d_in[0];
    const float* Wk   = (const float*)d_in[1];
    const float* bk   = (const float*)d_in[2];
    const float* Wq   = (const float*)d_in[3];
    const float* bq   = (const float*)d_in[4];
    const float* Wv   = (const float*)d_in[5];
    const float* bv   = (const float*)d_in[6];
    const float* kn_w = (const float*)d_in[7];
    const float* kn_b = (const float*)d_in[8];
    const float* qn_w = (const float*)d_in[9];
    const float* qn_b = (const float*)d_in[10];
    const float* vn_w = (const float*)d_in[11];
    const float* vn_b = (const float*)d_in[12];
    const float* W1   = (const float*)d_in[13];
    const float* b1   = (const float*)d_in[14];
    const float* W2   = (const float*)d_in[15];
    const float* b2   = (const float*)d_in[16];

    float* y_out = (float*)d_out;              // [256][256]
    float* A_out = (float*)d_out + 65536;      // [256][8][128][128]

    const size_t MATSZ = (size_t)32768 * 512;  // elems per projection matrix
    float*    proj = (float*)d_ws;                                         // 3*MATSZ f32
    _Float16* qkv  = (_Float16*)((char*)d_ws + 3 * MATSZ * sizeof(float)); // 3*MATSZ f16
    float* E = proj;               // reuse K-raw region after headnorm
    float* F = proj + MATSZ;       // reuse Q-raw region
    float* G = proj + 2 * MATSZ;   // reuse V-raw region (131072 floats used)

    dim3 gblk(128, 8);   // 32768/256 row tiles x 512/64 col tiles
    gemm_xw_kernel<<<gblk, 256, 0, stream>>>(x, Wk, bk, proj + 0 * MATSZ, 32768, 512, 500, 0);
    gemm_xw_kernel<<<gblk, 256, 0, stream>>>(x, Wq, bq, proj + 1 * MATSZ, 32768, 512, 500, 0);
    gemm_xw_kernel<<<gblk, 256, 0, stream>>>(x, Wv, bv, proj + 2 * MATSZ, 32768, 512, 500, 0);

    headnorm_kernel<<<6144, 256, 0, stream>>>(proj, kn_w, kn_b, qn_w, qn_b, vn_w, vn_b, qkv);

    attn_kernel<<<2048, 256, 0, stream>>>(qkv, A_out, E);

    gemm_xw_kernel<<<gblk, 256, 0, stream>>>(E, W1, b1, F, 32768, 512, 512, 1);

    finalnorm_maxpool_kernel<<<256, 256, 0, stream>>>(F, G);
    head_out_kernel<<<256, 256, 0, stream>>>(G, W2, b2, y_out);
}
